// CausalSelfAttention_64364379898244
// MI455X (gfx1250) — matrix operand order repaired and checked by reading, not compiled, not run on hardware
//
#include <hip/hip_runtime.h>
#include <hip/hip_bf16.h>

typedef __attribute__((ext_vector_type(16))) _Float16 v16h;
typedef __attribute__((ext_vector_type(8)))  _Float16 v8h;
typedef __attribute__((ext_vector_type(4)))  _Float16 v4h;
typedef __attribute__((ext_vector_type(8)))  float    v8f;

static constexpr int Bb = 2;
static constexpr int Tt = 2048;
static constexpr int Cc = 768;
static constexpr int Hh = 12;
static constexpr int Dd = 64;
static constexpr size_t QKV_ELEMS = (size_t)Bb * Hh * Tt * Dd;   // 3145728
static constexpr size_t X_ELEMS   = (size_t)Bb * Tt * Cc;        // 3145728
static constexpr size_t W_ELEMS   = (size_t)Cc * Cc;             // 589824
static constexpr float  NEG_BIG   = -1e30f;

__device__ __forceinline__ v8f wmma16(v16h a, v16h b, v8f c) {
  return __builtin_amdgcn_wmma_f32_16x16x32_f16(false, a, false, b, (short)0, c,
                                                false, false);
}

// ---- A-operand (16x32 f16, ISA 7.12.2): element i -> K = (i>>3)*16 + (lane>>4)*8 + (i&7)
// caller passes pointer at [row, kbase + (lane>>4)*8]; needs p[0..7] and p[16..23]
__device__ __forceinline__ v16h fragA_f16(const _Float16* p) {
  v8h lo = *(const v8h*)(p);
  v8h hi = *(const v8h*)(p + 16);
  v16h a;
#pragma unroll
  for (int i = 0; i < 8; ++i) { a[i] = lo[i]; a[8 + i] = hi[i]; }
  return a;
}
__device__ __forceinline__ v16h fragA_f32(const float* p) {
  v16h a;
#pragma unroll
  for (int i = 0; i < 8; ++i) a[i] = (_Float16)p[i];
#pragma unroll
  for (int i = 0; i < 8; ++i) a[8 + i] = (_Float16)p[16 + i];
  return a;
}
// ---- B-operand (32x16 f16): col n = lane&15, element i -> K = (i>>3)*16 + (lane>>4)*8 + (i&7).
// caller passes pointer at [n_row, kbase + (lane>>4)*8]; needs p[0..7] and p[16..23]
__device__ __forceinline__ v16h fragB_f16(const _Float16* p) {
  v8h lo = *(const v8h*)(p);
  v8h hi = *(const v8h*)(p + 16);
  v16h b;
#pragma unroll
  for (int i = 0; i < 8; ++i) { b[i] = lo[i]; b[8 + i] = hi[i]; }
  return b;
}

// ============================================================================
// Kernel 0: one-time f32 -> f16 conversion (vectorized, 4 elems/thread).
// ============================================================================
__global__ void __launch_bounds__(256)
cvt_kernel(const float* __restrict__ src, _Float16* __restrict__ dst, int n4) {
  const int i = blockIdx.x * 256 + threadIdx.x;
  if (i < n4) {
    const float4 f = ((const float4*)src)[i];
    v4h h;
    h[0] = (_Float16)f.x; h[1] = (_Float16)f.y;
    h[2] = (_Float16)f.z; h[3] = (_Float16)f.w;
    ((v4h*)dst)[i] = h;
  }
}

// ============================================================================
// Kernel 1: QKV projections.  O = X (4096x768) @ W^T (f16 operands).
// Block 256 thr = 8 waves (2x4), block tile 64x256, wave tile 32x64 (2x4 frags).
// Q,K stored (B,H,T,D) f16; V stored transposed (B,H,D,T) f16.
// ============================================================================
__global__ void __launch_bounds__(256)
qkv_kernel(const _Float16* __restrict__ Xh, const _Float16* __restrict__ Wqh,
           const _Float16* __restrict__ Wkh, const _Float16* __restrict__ Wvh,
           _Float16* __restrict__ Q, _Float16* __restrict__ K,
           _Float16* __restrict__ V) {
  const int lane = threadIdx.x & 31;
  const int wid  = threadIdx.x >> 5;
  const int mat  = blockIdx.z;
  const _Float16* W = (mat == 0) ? Wqh : (mat == 1) ? Wkh : Wvh;
  const int M0 = blockIdx.y * 64  + (wid >> 2) * 32;
  const int N0 = blockIdx.x * 256 + (wid & 3) * 64;
  const int lm  = lane & 15;
  const int g8  = (lane >> 4) * 8;
  const int g16 = (lane >> 4) * 16;

  v8f c[2][4] = {};
  for (int k = 0; k < Cc; k += 32) {
    v16h a0 = fragA_f16(Xh + (size_t)(M0 + lm) * Cc + k + g8);
    v16h a1 = fragA_f16(Xh + (size_t)(M0 + 16 + lm) * Cc + k + g8);
#pragma unroll
    for (int fn = 0; fn < 4; ++fn) {
      v16h b = fragB_f16(W + (size_t)(N0 + fn * 16 + lm) * Cc + k + g8);
      c[0][fn] = wmma16(a0, b, c[0][fn]);
      c[1][fn] = wmma16(a1, b, c[1][fn]);
    }
  }
#pragma unroll
  for (int fm = 0; fm < 2; ++fm)
#pragma unroll
    for (int fn = 0; fn < 4; ++fn)
#pragma unroll
      for (int j = 0; j < 8; ++j) {
        const int m = M0 + fm * 16 + g8 + j;
        const int n = N0 + fn * 16 + lm;
        const int b = m >> 11, t = m & (Tt - 1);
        const int h = n >> 6,  d = n & (Dd - 1);
        const _Float16 val = (_Float16)c[fm][fn][j];
        if (mat == 0)      Q[((size_t)(b * Hh + h) * Tt + t) * Dd + d] = val;
        else if (mat == 1) K[((size_t)(b * Hh + h) * Tt + t) * Dd + d] = val;
        else               V[((size_t)(b * Hh + h) * Dd + d) * Tt + t] = val;
      }
}

// ============================================================================
// Kernel 2: in-place RoPE on Q and K (first 32 dims, pairs d / d+16).
// ============================================================================
__global__ void __launch_bounds__(256)
rope_kernel(_Float16* __restrict__ Q, _Float16* __restrict__ K) {
  const int idx = blockIdx.x * 256 + threadIdx.x;   // 2 * 49152 * 16 threads
  const int d   = idx & 15;
  const int row = (idx >> 4);                        // 0 .. 2*49152-1
  const int NROWS = Bb * Hh * Tt;                    // 49152
  _Float16* base = (row >= NROWS) ? K : Q;
  const int r = (row >= NROWS) ? row - NROWS : row;  // (b*H+h)*T + t
  const int t = r & (Tt - 1);
  _Float16* u = base + (size_t)r * Dd;
  const float lnk = 0.28782313662425572f;            // ln(10000)/32
  const float ft = (float)t;
  const float f0 = ft * __expf(-lnk * (float)d);
  const float f1 = ft * __expf(-lnk * (float)(d + 16));
  const float u0 = (float)u[d];
  const float u1 = (float)u[d + 16];
  u[d]      = (_Float16)(u0 * __cosf(f0) - u1 * __sinf(f0));
  u[d + 16] = (_Float16)(u1 * __cosf(f1) + u0 * __sinf(f1));
}

// ============================================================================
// Kernel 3: raw scores S = 0.125 * Q K^T + rel_bias, causal mask -> att region.
// Block tile 64(q) x 256(k), wave tile 32x64.  Fully-masked blocks skip WMMA.
// ============================================================================
__global__ void __launch_bounds__(256)
scores_kernel(const _Float16* __restrict__ Q, const _Float16* __restrict__ K,
              const float* __restrict__ rel_bias, float* __restrict__ att) {
  const int lane = threadIdx.x & 31;
  const int wid  = threadIdx.x >> 5;
  const int bh   = blockIdx.z;
  const int h    = bh % Hh;
  const int M0 = blockIdx.y * 64  + (wid >> 2) * 32;   // q
  const int N0 = blockIdx.x * 256 + (wid & 3) * 64;    // k
  const int lm  = lane & 15;
  const int g8  = (lane >> 4) * 8;
  const int g16 = (lane >> 4) * 16;
  const _Float16* Qb = Q + (size_t)bh * Tt * Dd;
  const _Float16* Kb = K + (size_t)bh * Tt * Dd;
  float* S = att + (size_t)bh * Tt * Tt;

  const bool anyValid = (blockIdx.y * 64 + 63) >= (int)(blockIdx.x * 256);
  v8f c[2][4] = {};
  if (anyValid) {
#pragma unroll
    for (int k = 0; k < Dd; k += 32) {
      v16h a0 = fragA_f16(Qb + (size_t)(M0 + lm) * Dd + k + g8);
      v16h a1 = fragA_f16(Qb + (size_t)(M0 + 16 + lm) * Dd + k + g8);
#pragma unroll
      for (int fn = 0; fn < 4; ++fn) {
        v16h b = fragB_f16(Kb + (size_t)(N0 + fn * 16 + lm) * Dd + k + g8);
        c[0][fn] = wmma16(a0, b, c[0][fn]);
        c[1][fn] = wmma16(a1, b, c[1][fn]);
      }
    }
  }
#pragma unroll
  for (int fm = 0; fm < 2; ++fm)
#pragma unroll
    for (int fn = 0; fn < 4; ++fn)
#pragma unroll
      for (int j = 0; j < 8; ++j) {
        const int q  = M0 + fm * 16 + g8 + j;
        const int kk = N0 + fn * 16 + lm;
        float s = NEG_BIG;
        if (kk <= q)
          s = c[fm][fn][j] * 0.125f + rel_bias[(q - kk + (Tt - 1)) * Hh + h];
        S[(size_t)q * Tt + kk] = s;
      }
}

// ============================================================================
// Kernel 4: in-place row softmax over att (one block per row of 2048).
// ============================================================================
__global__ void __launch_bounds__(256)
softmax_kernel(float* __restrict__ att) {
  __shared__ float red[256];
  const int tid = threadIdx.x;
  float* row = att + (size_t)blockIdx.x * Tt;
  float v[8];
  float m = NEG_BIG;
#pragma unroll
  for (int i = 0; i < 8; ++i) { v[i] = row[tid + i * 256]; m = fmaxf(m, v[i]); }
  red[tid] = m;
  __syncthreads();
  for (int s = 128; s > 0; s >>= 1) {
    if (tid < s) red[tid] = fmaxf(red[tid], red[tid + s]);
    __syncthreads();
  }
  m = red[0];
  __syncthreads();
  float sum = 0.f;
#pragma unroll
  for (int i = 0; i < 8; ++i) { v[i] = __expf(v[i] - m); sum += v[i]; }
  red[tid] = sum;
  __syncthreads();
  for (int s = 128; s > 0; s >>= 1) {
    if (tid < s) red[tid] += red[tid + s];
    __syncthreads();
  }
  const float inv = 1.0f / red[0];
#pragma unroll
  for (int i = 0; i < 8; ++i) row[tid + i * 256] = v[i] * inv;
}

// ============================================================================
// Kernel 5: yh = att @ V.  Block tile 256(q) x 64(d), 8 waves along M,
// wave tile 32x64 (2x4 frags).  att loaded f32->f16 on the fly (read-once);
// Vt is (B,H,D,T) so B-operand loads are contiguous.
// Causal: k-loop only up to the block's highest query row.
// yh written (B*T, C) row-major f16.
// ============================================================================
__global__ void __launch_bounds__(256)
av_kernel(const float* __restrict__ att, const _Float16* __restrict__ Vt,
          _Float16* __restrict__ YH) {
  const int lane = threadIdx.x & 31;
  const int wid  = threadIdx.x >> 5;
  const int bh   = blockIdx.y;
  const int b = bh / Hh, h = bh % Hh;
  const float*    A  = att + (size_t)bh * Tt * Tt;
  const _Float16* Vb = Vt  + (size_t)bh * Dd * Tt;
  const int M0 = blockIdx.x * 256 + wid * 32;          // q rows
  const int lm  = lane & 15;
  const int g8  = (lane >> 4) * 8;
  const int g16 = (lane >> 4) * 16;

  v8f c[2][4] = {};
  const int kmax = blockIdx.x * 256 + 255;             // highest valid k (<= 2047)
  for (int kt = 0; kt <= kmax; kt += 32) {
    v16h a0 = fragA_f32(A + (size_t)(M0 + lm) * Tt + kt + g8);
    v16h a1 = fragA_f32(A + (size_t)(M0 + 16 + lm) * Tt + kt + g8);
#pragma unroll
    for (int fn = 0; fn < 4; ++fn) {
      v16h bfrag = fragB_f16(Vb + (size_t)(fn * 16 + lm) * Tt + kt + g8);
      c[0][fn] = wmma16(a0, bfrag, c[0][fn]);
      c[1][fn] = wmma16(a1, bfrag, c[1][fn]);
    }
  }
#pragma unroll
  for (int fm = 0; fm < 2; ++fm)
#pragma unroll
    for (int fn = 0; fn < 4; ++fn)
#pragma unroll
      for (int j = 0; j < 8; ++j) {
        const int q = M0 + fm * 16 + g8 + j;
        const int d = fn * 16 + lm;
        YH[((size_t)(b * Tt + q)) * Cc + h * Dd + d] = (_Float16)c[fm][fn][j];
      }
}

// ============================================================================
// Kernel 6: y = yh @ Wp^T + bp.  Same tiling as kernel 1, fp32 output.
// ============================================================================
__global__ void __launch_bounds__(256)
proj_kernel(const _Float16* __restrict__ YH, const _Float16* __restrict__ Wph,
            const float* __restrict__ bp, float* __restrict__ y) {
  const int lane = threadIdx.x & 31;
  const int wid  = threadIdx.x >> 5;
  const int M0 = blockIdx.y * 64  + (wid >> 2) * 32;
  const int N0 = blockIdx.x * 256 + (wid & 3) * 64;
  const int lm  = lane & 15;
  const int g8  = (lane >> 4) * 8;
  const int g16 = (lane >> 4) * 16;

  v8f c[2][4] = {};
  for (int k = 0; k < Cc; k += 32) {
    v16h a0 = fragA_f16(YH + (size_t)(M0 + lm) * Cc + k + g8);
    v16h a1 = fragA_f16(YH + (size_t)(M0 + 16 + lm) * Cc + k + g8);
#pragma unroll
    for (int fn = 0; fn < 4; ++fn) {
      v16h b = fragB_f16(Wph + (size_t)(N0 + fn * 16 + lm) * Cc + k + g8);
      c[0][fn] = wmma16(a0, b, c[0][fn]);
      c[1][fn] = wmma16(a1, b, c[1][fn]);
    }
  }
#pragma unroll
  for (int fm = 0; fm < 2; ++fm)
#pragma unroll
    for (int fn = 0; fn < 4; ++fn)
#pragma unroll
      for (int j = 0; j < 8; ++j) {
        const int m = M0 + fm * 16 + g8 + j;
        const int n = N0 + fn * 16 + lm;
        y[(size_t)m * Cc + n] = c[fm][fn][j] + bp[n];
      }
}

// ============================================================================
extern "C" void kernel_launch(void* const* d_in, const int* in_sizes, int n_in,
                              void* d_out, int out_size, void* d_ws, size_t ws_size,
                              hipStream_t stream) {
  (void)in_sizes; (void)n_in; (void)out_size; (void)ws_size;
  const float* x        = (const float*)d_in[0];
  const float* Wq       = (const float*)d_in[1];
  const float* Wk       = (const float*)d_in[2];
  const float* Wv       = (const float*)d_in[3];
  const float* Wp       = (const float*)d_in[4];
  const float* bp       = (const float*)d_in[5];
  const float* rel_bias = (const float*)d_in[6];

  float* y   = (float*)d_out;
  float* att = y + (size_t)Bb * Tt * Cc;             // att output region

  _Float16* Q   = (_Float16*)d_ws;                   // (B,H,T,D)
  _Float16* K   = Q   + QKV_ELEMS;                   // (B,H,T,D)
  _Float16* Vt  = K   + QKV_ELEMS;                   // (B,H,D,T)
  _Float16* YH  = Vt  + QKV_ELEMS;                   // (B*T, C)
  _Float16* Xh  = YH  + X_ELEMS;                     // (B*T, C) f16 copy of x
  _Float16* Wqh = Xh  + X_ELEMS;
  _Float16* Wkh = Wqh + W_ELEMS;
  _Float16* Wvh = Wkh + W_ELEMS;
  _Float16* Wph = Wvh + W_ELEMS;

  // 0) one-time f32 -> f16 conversions
  cvt_kernel<<<(int)(X_ELEMS / 4 + 255) / 256, 256, 0, stream>>>(x,  Xh,  (int)(X_ELEMS / 4));
  cvt_kernel<<<(int)(W_ELEMS / 4 + 255) / 256, 256, 0, stream>>>(Wq, Wqh, (int)(W_ELEMS / 4));
  cvt_kernel<<<(int)(W_ELEMS / 4 + 255) / 256, 256, 0, stream>>>(Wk, Wkh, (int)(W_ELEMS / 4));
  cvt_kernel<<<(int)(W_ELEMS / 4 + 255) / 256, 256, 0, stream>>>(Wv, Wvh, (int)(W_ELEMS / 4));
  cvt_kernel<<<(int)(W_ELEMS / 4 + 255) / 256, 256, 0, stream>>>(Wp, Wph, (int)(W_ELEMS / 4));

  // 1) QKV projections (WMMA f16, fp32 accum)
  qkv_kernel<<<dim3(Cc / 256, (Bb * Tt) / 64, 3), 256, 0, stream>>>(
      Xh, Wqh, Wkh, Wvh, Q, K, Vt);
  // 2) RoPE in place on Q,K
  rope_kernel<<<(2 * Bb * Hh * Tt * 16) / 256, 256, 0, stream>>>(Q, K);
  // 3) raw scores + bias + mask -> att region
  scores_kernel<<<dim3(Tt / 256, Tt / 64, Bb * Hh), 256, 0, stream>>>(
      Q, K, rel_bias, att);
  // 4) in-place softmax
  softmax_kernel<<<Bb * Hh * Tt, 256, 0, stream>>>(att);
  // 5) att @ V -> yh
  av_kernel<<<dim3(Tt / 256, Bb * Hh), 256, 0, stream>>>(att, Vt, YH);
  // 6) output projection
  proj_kernel<<<dim3(Cc / 256, (Bb * Tt) / 64), 256, 0, stream>>>(YH, Wph, bp, y);
}
